// Attention_65876208386716
// MI455X (gfx1250) — matrix operand order repaired and checked by reading, not compiled, not run on hardware
//
#include <hip/hip_runtime.h>

// ---------------- problem constants ----------------
#define DIMX 1024
#define HEADS 16
#define HEAD_DIM 64
#define SEQ 2048
#define BATCH 4
#define MTOT (BATCH * SEQ)      // 8192 tokens
#define NQKV (3 * DIMX)         // 3072
#define ATTN_SCALE 0.125f       // 64^-0.5

// ---------------- WMMA fragment plumbing ----------------
typedef __attribute__((ext_vector_type(16))) __bf16 v16bf;
typedef __attribute__((ext_vector_type(8)))  float  v8f;

union FragBF { v16bf v; uint4 u[2]; };

// A-matrix 16x32 bf16 (ISA 7.12.2): lane<16 -> row=lane, K {0..7,16..23};
// lane>=16 -> row=lane-16, K {8..15,24..31}.  Source is row-major, ld elems.
__device__ __forceinline__ FragBF load_a_frag(const __bf16* p, int ld, int row,
                                              int col, int half) {
  FragBF f;
  const __bf16* base = p + (long)row * ld + col + half * 8;
  f.u[0] = *reinterpret_cast<const uint4*>(base);
  f.u[1] = *reinterpret_cast<const uint4*>(base + 16);
  return f;
}

// B-matrix 32x16 bf16: lane<16 -> col N=lane, K {0..7,16..23};
// lane>=16 -> col N=lane-16, K {8..15,24..31}.  Source is B^T row-major (row = N).
__device__ __forceinline__ FragBF load_b_frag(const __bf16* p, int ld, int row,
                                              int col, int half) {
  FragBF f;
  const __bf16* base = p + (long)row * ld + col + half * 8;
  f.u[0] = *reinterpret_cast<const uint4*>(base);
  f.u[1] = *reinterpret_cast<const uint4*>(base + 16);
  return f;
}

__device__ __forceinline__ v8f wmma_bf16(const FragBF& a, const FragBF& b, v8f c) {
  return __builtin_amdgcn_wmma_f32_16x16x32_bf16(false, a.v, false, b.v,
                                                 (short)0, c, false, false);
}

// ---------------- prepass: conversions ----------------
__global__ __launch_bounds__(256) void cvt_f32_bf16_kernel(
    const float* __restrict__ in, __bf16* __restrict__ out, int n) {
  int i = (blockIdx.x * 256 + threadIdx.x) * 4;
  if (i < n) {
    float4 f = *reinterpret_cast<const float4*>(in + i);
    out[i + 0] = (__bf16)f.x;
    out[i + 1] = (__bf16)f.y;
    out[i + 2] = (__bf16)f.z;
    out[i + 3] = (__bf16)f.w;
  }
}

// in [rows][cols] f32  ->  out [cols][rows] bf16 (coalesced reads)
__global__ __launch_bounds__(256) void transpose_f32_bf16_kernel(
    const float* __restrict__ in, __bf16* __restrict__ out, int rows, int cols) {
  int idx = blockIdx.x * 256 + threadIdx.x;
  if (idx < rows * cols) {
    int r = idx / cols, c = idx - r * cols;
    out[(long)c * rows + r] = (__bf16)in[idx];
  }
}

// ---------------- QKV GEMM: [8192,1024] x [1024,3072] ----------------
// block = 256 thr = 8 waves (2 M x 4 N); wave tile 32x64; block tile 64x256.
__global__ __launch_bounds__(256) void qkv_gemm_kernel(
    const __bf16* __restrict__ X, const __bf16* __restrict__ Wt,
    __bf16* __restrict__ Q, __bf16* __restrict__ K, __bf16* __restrict__ Vt) {
  const int lane = threadIdx.x & 31, wid = threadIdx.x >> 5;
  const int half = lane >> 4, lrow = lane & 15;
  const int m0 = blockIdx.x * 64 + (wid & 1) * 32;
  const int n0 = blockIdx.y * 256 + (wid >> 1) * 64;

  v8f acc[2][4];
#pragma unroll
  for (int i = 0; i < 2; ++i)
#pragma unroll
    for (int j = 0; j < 4; ++j) acc[i][j] = (v8f)(0.0f);

  for (int k = 0; k < DIMX; k += 32) {
    FragBF a[2], b[4];
#pragma unroll
    for (int i = 0; i < 2; ++i) a[i] = load_a_frag(X, DIMX, m0 + 16 * i + lrow, k, half);
#pragma unroll
    for (int j = 0; j < 4; ++j) b[j] = load_b_frag(Wt, DIMX, n0 + 16 * j + lrow, k, half);
#pragma unroll
    for (int i = 0; i < 2; ++i)
#pragma unroll
      for (int j = 0; j < 4; ++j) acc[i][j] = wmma_bf16(a[i], b[j], acc[i][j]);
  }

  // Scatter epilogue: n -> {q,k,v}, head, dim.  C layout: row=r+8*half, col=lrow.
#pragma unroll
  for (int j = 0; j < 4; ++j) {
    const int n = n0 + 16 * j + lrow;
    const int which = n >> 10;          // 0=q 1=k 2=v
    const int h = (n & 1023) >> 6;
    const int d = n & 63;
#pragma unroll
    for (int i = 0; i < 2; ++i)
#pragma unroll
      for (int r = 0; r < 8; ++r) {
        const int m = m0 + 16 * i + r + 8 * half;
        const int bidx = m >> 11, tok = m & (SEQ - 1);
        const long bh = (long)bidx * HEADS + h;
        const float val = acc[i][j][r];
        if (which == 0)      Q[(bh * SEQ + tok) * HEAD_DIM + d] = (__bf16)(val * ATTN_SCALE);
        else if (which == 1) K[(bh * SEQ + tok) * HEAD_DIM + d] = (__bf16)val;
        else                 Vt[(bh * HEAD_DIM + d) * SEQ + tok] = (__bf16)val;
      }
  }
}

// ---------------- flash attention ----------------
// grid (SEQ/128, B*H); 8 waves/block, each wave owns 16 q-rows.
__global__ __launch_bounds__(256) void flash_attn_kernel(
    const __bf16* __restrict__ Q, const __bf16* __restrict__ K,
    const __bf16* __restrict__ Vt, __bf16* __restrict__ O) {
  __shared__ __bf16 lds_p[8][16 * 32];   // per-wave 16x32 P tile (C->A transpose)

  const int lane = threadIdx.x & 31, wid = threadIdx.x >> 5;
  const int half = lane >> 4, lrow = lane & 15;
  const int bh = blockIdx.y;
  const int q0 = blockIdx.x * 128 + wid * 16;

  const __bf16* Qb = Q + (long)bh * SEQ * HEAD_DIM;
  const __bf16* Kb = K + (long)bh * SEQ * HEAD_DIM;
  const __bf16* Vb = Vt + (long)bh * HEAD_DIM * SEQ;

  FragBF qf[2];
#pragma unroll
  for (int c = 0; c < 2; ++c) qf[c] = load_a_frag(Qb, HEAD_DIM, q0 + lrow, 32 * c, half);

  v8f o[4];
#pragma unroll
  for (int t = 0; t < 4; ++t) o[t] = (v8f)(0.0f);
  float mrow[8], lrowsum[8];
#pragma unroll
  for (int r = 0; r < 8; ++r) { mrow[r] = -1e30f; lrowsum[r] = 0.0f; }

  for (int kb = 0; kb < SEQ; kb += 32) {
    // S = Q * K^T for keys [kb, kb+32): two 16x16 C tiles
    v8f s0 = (v8f)(0.0f), s1 = (v8f)(0.0f);
#pragma unroll
    for (int c = 0; c < 2; ++c) {
      FragBF k0 = load_b_frag(Kb, HEAD_DIM, kb + lrow, 32 * c, half);
      FragBF k1 = load_b_frag(Kb, HEAD_DIM, kb + 16 + lrow, 32 * c, half);
      s0 = wmma_bf16(qf[c], k0, s0);
      s1 = wmma_bf16(qf[c], k1, s1);
    }

    // online softmax; each row lives across a 16-lane group in one VGPR
    float p0[8], p1[8];
#pragma unroll
    for (int r = 0; r < 8; ++r) {
      float mx = fmaxf(s0[r], s1[r]);
#pragma unroll
      for (int off = 8; off >= 1; off >>= 1) mx = fmaxf(mx, __shfl_xor(mx, off, 32));
      const float mn = fmaxf(mrow[r], mx);
      const float sc = __expf(mrow[r] - mn);
      p0[r] = __expf(s0[r] - mn);
      p1[r] = __expf(s1[r] - mn);
      float rs = p0[r] + p1[r];
#pragma unroll
      for (int off = 8; off >= 1; off >>= 1) rs += __shfl_xor(rs, off, 32);
      lrowsum[r] = lrowsum[r] * sc + rs;
      mrow[r] = mn;
#pragma unroll
      for (int t = 0; t < 4; ++t) o[t][r] *= sc;
    }

    // C-layout P -> LDS row-major [16 q][32 keys], reload as A fragment.
    // Same-wave DS ops are hardware in-order; compiler barrier keeps order.
    __bf16* pb = lds_p[wid];
#pragma unroll
    for (int r = 0; r < 8; ++r) {
      const int row = r + 8 * half;
      pb[row * 32 + lrow]      = (__bf16)p0[r];
      pb[row * 32 + 16 + lrow] = (__bf16)p1[r];
    }
    asm volatile("" ::: "memory");
    FragBF pf;
    pf.u[0] = *reinterpret_cast<const uint4*>(pb + lrow * 32 + half * 8);
    pf.u[1] = *reinterpret_cast<const uint4*>(pb + lrow * 32 + 16 + half * 8);
    asm volatile("" ::: "memory");

    // O += P * V  (V stored transposed [d][n] -> contiguous B rows)
#pragma unroll
    for (int t = 0; t < 4; ++t) {
      FragBF vf = load_b_frag(Vb, SEQ, 16 * t + lrow, kb, half);
      o[t] = wmma_bf16(pf, vf, o[t]);
    }
  }

  // epilogue: normalize and store O as [B, tok, h*64+d] bf16
  const int b = bh >> 4, h = bh & 15;
#pragma unroll
  for (int r = 0; r < 8; ++r) {
    const float inv = 1.0f / lrowsum[r];
    const int tok = q0 + r + 8 * half;
#pragma unroll
    for (int t = 0; t < 4; ++t)
      O[((long)b * SEQ + tok) * DIMX + h * 64 + 16 * t + lrow] = (__bf16)(o[t][r] * inv);
  }
}

// ---------------- output projection GEMM + bias ----------------
__global__ __launch_bounds__(256) void proj_gemm_kernel(
    const __bf16* __restrict__ Oin, const __bf16* __restrict__ Wt,
    const float* __restrict__ bias, float* __restrict__ out) {
  const int lane = threadIdx.x & 31, wid = threadIdx.x >> 5;
  const int half = lane >> 4, lrow = lane & 15;
  const int m0 = blockIdx.x * 64 + (wid & 1) * 32;
  const int n0 = blockIdx.y * 256 + (wid >> 1) * 64;

  v8f acc[2][4];
#pragma unroll
  for (int i = 0; i < 2; ++i)
#pragma unroll
    for (int j = 0; j < 4; ++j) acc[i][j] = (v8f)(0.0f);

  for (int k = 0; k < DIMX; k += 32) {
    FragBF a[2], b[4];
#pragma unroll
    for (int i = 0; i < 2; ++i) a[i] = load_a_frag(Oin, DIMX, m0 + 16 * i + lrow, k, half);
#pragma unroll
    for (int j = 0; j < 4; ++j) b[j] = load_b_frag(Wt, DIMX, n0 + 16 * j + lrow, k, half);
#pragma unroll
    for (int i = 0; i < 2; ++i)
#pragma unroll
      for (int j = 0; j < 4; ++j) acc[i][j] = wmma_bf16(a[i], b[j], acc[i][j]);
  }

#pragma unroll
  for (int j = 0; j < 4; ++j) {
    const int n = n0 + 16 * j + lrow;
    const float bv = bias[n];
#pragma unroll
    for (int i = 0; i < 2; ++i)
#pragma unroll
      for (int r = 0; r < 8; ++r) {
        const int m = m0 + 16 * i + r + 8 * half;
        out[(long)m * DIMX + n] = acc[i][j][r] + bv;
      }
  }
}

// ---------------- host launch ----------------
extern "C" void kernel_launch(void* const* d_in, const int* in_sizes, int n_in,
                              void* d_out, int out_size, void* d_ws, size_t ws_size,
                              hipStream_t stream) {
  const float* x      = (const float*)d_in[0];  // [4,2048,1024]
  const float* w_qkv  = (const float*)d_in[1];  // [1024,3072]
  const float* w_proj = (const float*)d_in[2];  // [1024,1024]
  const float* b_proj = (const float*)d_in[3];  // [1024]
  float* out = (float*)d_out;

  char* ws = (char*)d_ws;                       // ~92.3 MB used
  __bf16* x_bf   = (__bf16*)(ws);                              // 16 MB
  __bf16* wqkvT  = (__bf16*)(ws + 16777216);                   //  6 MB
  __bf16* wprojT = (__bf16*)(ws + 23068672);                   //  2 MB
  __bf16* Qb     = (__bf16*)(ws + 25165824);                   // 16 MB
  __bf16* Kb     = (__bf16*)(ws + 41943040);                   // 16 MB
  __bf16* Vt     = (__bf16*)(ws + 58720256);                   // 16 MB
  __bf16* Ob     = (__bf16*)(ws + 75497472);                   // 16 MB

  // 1. conversions / weight transposes
  cvt_f32_bf16_kernel<<<(MTOT * DIMX / 4 + 255) / 256, 256, 0, stream>>>(
      x, x_bf, MTOT * DIMX);
  transpose_f32_bf16_kernel<<<(DIMX * NQKV + 255) / 256, 256, 0, stream>>>(
      w_qkv, wqkvT, DIMX, NQKV);
  transpose_f32_bf16_kernel<<<(DIMX * DIMX + 255) / 256, 256, 0, stream>>>(
      w_proj, wprojT, DIMX, DIMX);

  // 2. fused QKV projection -> Q (pre-scaled), K, V^T
  qkv_gemm_kernel<<<dim3(MTOT / 64, NQKV / 256), 256, 0, stream>>>(
      x_bf, wqkvT, Qb, Kb, Vt);

  // 3. flash attention
  flash_attn_kernel<<<dim3(SEQ / 128, BATCH * HEADS), 256, 0, stream>>>(
      Qb, Kb, Vt, Ob);

  // 4. output projection + bias
  proj_gemm_kernel<<<dim3(MTOT / 64, DIMX / 256), 256, 0, stream>>>(
      Ob, wprojT, b_proj, out);
}
